// SAMGalaxyGNN_MultiZ_40355512713546
// MI455X (gfx1250) — matrix operand order repaired and checked by reading, not compiled, not run on hardware
//
#include <hip/hip_runtime.h>
#include <hip/hip_bf16.h>

typedef __bf16 bf16_t;
typedef bf16_t v16bf __attribute__((ext_vector_type(16)));
typedef float  v8f   __attribute__((ext_vector_type(8)));

#if defined(__gfx1250__) && __has_builtin(__builtin_amdgcn_global_load_async_to_lds_b128) && __has_builtin(__builtin_amdgcn_s_wait_asynccnt)
#define USE_ASYNC_LDS 1
#endif

#ifdef USE_ASYNC_LDS
typedef int async_i4 __attribute__((vector_size(16)));
__device__ __forceinline__ void async_copy_b128(const float* g, float* l) {
  __builtin_amdgcn_global_load_async_to_lds_b128(
      (__attribute__((address_space(1))) async_i4*)g,
      (__attribute__((address_space(3))) async_i4*)l, 0, 0);
}
#endif

__device__ __forceinline__ v8f zero_v8f() {
  v8f a;
#pragma unroll
  for (int i = 0; i < 8; ++i) a[i] = 0.0f;
  return a;
}

__device__ __forceinline__ v8f wmma_bf16(v16bf a, v16bf b, v8f c) {
  // (neg_a, A, neg_b, B, c_mod, C, reuse_a, reuse_b)
  return __builtin_amdgcn_wmma_f32_16x16x32_bf16(false, a, false, b, (short)0, c, false, false);
}

// Build 16x32 bf16 A-fragment from an LDS fp32 tile (row-major, leading dim lda).
// lanes 0-15 -> M=lane, K-octet base 0 ; lanes 16-31 -> M=lane-16, K-octet base 8.
// VGPR 0..3 : K = kb+0..7 ; VGPR 4..7 : K = 16+kb+0..7.
__device__ __forceinline__ v16bf make_a_frag(const float* __restrict__ buf, int lda, int kc, int lane) {
  const int m  = lane & 15;
  const int kb = (lane >> 4) << 3;
  const float* r = buf + m * lda + kc;
  v16bf a;
#pragma unroll
  for (int e = 0; e < 8; ++e) a[e] = (bf16_t)r[kb + e];
#pragma unroll
  for (int e = 0; e < 8; ++e) a[8 + e] = (bf16_t)r[16 + kb + e];
  return a;
}

// Packed-fragment load: fragment (t,k) of a pre-packed bf16 weight, 32 B per lane.
__device__ __forceinline__ v16bf load_bfrag(const bf16_t* __restrict__ P, int t, int k, int nk, int lane) {
  return *(const v16bf*)(P + ((size_t)(t * nk + k) * 32 + lane) * 16);
}

__device__ __forceinline__ float gelu_exact(float x) {
  return 0.5f * x * (1.0f + erff(x * 0.70710678118654752f));
}
__device__ __forceinline__ float lrelu(float x) { return x > 0.0f ? x : 0.01f * x; }

// ------------------------------------------------------------ weight packing
// Pack W[z][nOut x K] (row stride ldw) into B-fragment-contiguous bf16 layout:
// P[((t*nk + k)*32 + lane)*16 + e] ; B-frag lane layout: n = t*16 + (lane&15),
// kc = k*32 + (lane>>4)*8, e = 0..7 -> K = kc+e ; e = 8..15 -> K = kc+16+(e-8).
__global__ void pack_b_kernel(const float* __restrict__ W, bf16_t* __restrict__ P,
                              int nOut, int K, int ldw, long long sWz, long long sPz) {
  const int nk = K >> 5;
  const int ntiles = (nOut + 15) >> 4;
  const int total = ntiles * nk * 32;
  const int idx = blockIdx.x * blockDim.x + threadIdx.x;
  if (idx >= total) return;
  const int lane = idx & 31;
  const int tk = idx >> 5;
  const int k = tk % nk;
  const int t = tk / nk;
  const float* Wz = W + (long long)blockIdx.y * sWz;
  bf16_t* Pz = P + (long long)blockIdx.y * sPz;
  int n = (t << 4) + (lane & 15);
  n = (n < nOut) ? n : (nOut - 1);
  const int kc = (k << 5) + ((lane >> 4) << 3);
  const float* r = Wz + (size_t)n * ldw + kc;
  bf16_t* o = Pz + ((size_t)(t * nk + k) * 32 + lane) * 16;
#pragma unroll
  for (int e = 0; e < 8; ++e) o[e] = (bf16_t)r[e];
#pragma unroll
  for (int e = 0; e < 8; ++e) o[8 + e] = (bf16_t)r[16 + e];
}

// ---------------------------------------------------------------- utilities
__global__ void zero_kernel(float* __restrict__ p, long long n) {
  long long i = (long long)blockIdx.x * blockDim.x + threadIdx.x;
  if (i < n) p[i] = 0.0f;
}

__global__ void degree_kernel(const int* __restrict__ ei, float* __restrict__ deg, int E) {
  int e = blockIdx.x * blockDim.x + threadIdx.x;
  if (e < E) atomicAdd(&deg[ei[E + e]], 1.0f);
}

__global__ void scatter_kernel(const float* __restrict__ h, const int* __restrict__ ei,
                               float* __restrict__ agg, int E, int Kin) {
  long long idx = (long long)blockIdx.x * blockDim.x + threadIdx.x;
  const int q = Kin >> 2;
  if (idx >= (long long)E * q) return;
  int e  = (int)(idx / q);
  int g4 = (int)(idx - (long long)e * q);
  int src = ei[e], dst = ei[E + e];
  const float4 v = *(const float4*)(h + (size_t)src * Kin + g4 * 4);
  float* p = agg + (size_t)dst * Kin + g4 * 4;
  atomicAdd(p + 0, v.x);
  atomicAdd(p + 1, v.y);
  atomicAdd(p + 2, v.z);
  atomicAdd(p + 3, v.w);
}

// u_emb = lrelu(u @ linu_W^T + linu_b); cvec[j] = lin1_b[j] + sum_k u_emb[k]*lin1_W[j][128+k]
__global__ void prep_kernel(const float* __restrict__ u, const float* __restrict__ linuW,
                            const float* __restrict__ linub, const float* __restrict__ lin1W,
                            const float* __restrict__ lin1b, float* __restrict__ cvec) {
  __shared__ float ue[128];
  int j = threadIdx.x;
  float s = linub[j];
#pragma unroll
  for (int k = 0; k < 16; ++k) s += u[k] * linuW[j * 16 + k];
  ue[j] = lrelu(s);
  __syncthreads();
  float t = lin1b[j];
  for (int k = 0; k < 128; ++k) t += ue[k] * lin1W[j * 256 + 128 + k];
  cvec[j] = t;
}

// --------------------------------------------------- SAGEConv + LN + GELU layer
template <int KIN>
__global__ __launch_bounds__(128) void gnn_layer_kernel(
    const float* __restrict__ hin, const float* __restrict__ agg, const float* __restrict__ deg,
    const bf16_t* __restrict__ pWl, const float* __restrict__ bl, const bf16_t* __restrict__ pWr,
    const float* __restrict__ lng, const float* __restrict__ lnb,
    float* __restrict__ hout, int N) {
  constexpr int NK = KIN / 32;
  __shared__ float sA[16 * KIN];
  __shared__ float sH[16 * KIN];
  __shared__ float sO[16 * 128];
  __shared__ float sDS[16];
  const int tid = threadIdx.x, wave = tid >> 5, lane = tid & 31;
  const int row0 = blockIdx.x * 16;

  if (tid < 16) {
    int node = row0 + tid;
    node = node < N ? node : N - 1;
    sDS[tid] = 1.0f / fmaxf(deg[node], 1.0f);
  }
  const int q = KIN >> 2;
  for (int c = tid; c < 16 * q; c += 128) {
    const int r = c / q, g4 = c - r * q;
    int node = row0 + r;
    node = node < N ? node : N - 1;
    const float* ga = agg + (size_t)node * KIN + g4 * 4;
    const float* gh = hin + (size_t)node * KIN + g4 * 4;
#ifdef USE_ASYNC_LDS
    async_copy_b128(ga, sA + r * KIN + g4 * 4);
    async_copy_b128(gh, sH + r * KIN + g4 * 4);
#else
    *(float4*)(sA + r * KIN + g4 * 4) = *(const float4*)ga;
    *(float4*)(sH + r * KIN + g4 * 4) = *(const float4*)gh;
#endif
  }
#ifdef USE_ASYNC_LDS
  __builtin_amdgcn_s_wait_asynccnt(0);
#endif
  __syncthreads();

  v16bf aA[NK], aH[NK];
#pragma unroll
  for (int k = 0; k < NK; ++k) {
    aA[k] = make_a_frag(sA, KIN, k << 5, lane);
    aH[k] = make_a_frag(sH, KIN, k << 5, lane);
  }
  const int mb = (lane >= 16) ? 8 : 0;
  const int nlo = lane & 15;
  for (int t = wave; t < 8; t += 4) {
    v8f accA = zero_v8f(), accH = zero_v8f();
#pragma unroll
    for (int k = 0; k < NK; ++k) {
      accA = wmma_bf16(aA[k], load_bfrag(pWl, t, k, NK, lane), accA);
      accH = wmma_bf16(aH[k], load_bfrag(pWr, t, k, NK, lane), accH);
    }
    const int col = (t << 4) + nlo;
    const float bb = bl[col];
#pragma unroll
    for (int r = 0; r < 8; ++r)
      sO[(mb + r) * 128 + col] = accA[r] * sDS[mb + r] + accH[r] + bb;  // agg/deg fold on D rows
  }
  __syncthreads();

  // LayerNorm + GELU: 16 rows x 8 threads
  const int r = tid >> 3, sub = tid & 7;
  float s = 0.0f, s2 = 0.0f;
  for (int cc = sub; cc < 128; cc += 8) {
    float v = sO[r * 128 + cc];
    s += v;
    s2 += v * v;
  }
#pragma unroll
  for (int o = 1; o < 8; o <<= 1) {
    s  += __shfl_xor(s, o, 32);
    s2 += __shfl_xor(s2, o, 32);
  }
  const float mu  = s * 0.0078125f;
  const float var = s2 * 0.0078125f - mu * mu;
  const float inv = rsqrtf(var + 1e-5f);
  const int orow = row0 + r;
  if (orow < N) {
    for (int cc = sub; cc < 128; cc += 8) {
      float v = (sO[r * 128 + cc] - mu) * inv * lng[cc] + lnb[cc];
      hout[(size_t)orow * 128 + cc] = gelu_exact(v);
    }
  }
}

// ---------------- hs = lrelu(h[yidx] @ lin1W[:, :128]^T + c), stored in bf16 A-frag layout
__global__ __launch_bounds__(128) void hs_kernel(
    const float* __restrict__ hin, const int* __restrict__ yidx,
    const bf16_t* __restrict__ pLin1, const float* __restrict__ cvec,
    bf16_t* __restrict__ hs_pk, int M) {
  __shared__ float sH[16 * 128];
  __shared__ float sO[16 * 128];
  const int tid = threadIdx.x, wave = tid >> 5, lane = tid & 31;
  const int m0 = blockIdx.x * 16;

  for (int c = tid; c < 16 * 32; c += 128) {
    const int r = c >> 5, g4 = c & 31;
    const int m = m0 + r;
    const int node = yidx[(m < M) ? m : (M - 1)];
    const float* gh = hin + (size_t)node * 128 + g4 * 4;
#ifdef USE_ASYNC_LDS
    async_copy_b128(gh, sH + r * 128 + g4 * 4);
#else
    *(float4*)(sH + r * 128 + g4 * 4) = *(const float4*)gh;
#endif
  }
#ifdef USE_ASYNC_LDS
  __builtin_amdgcn_s_wait_asynccnt(0);
#endif
  __syncthreads();

  v16bf a[4];
#pragma unroll
  for (int k = 0; k < 4; ++k) a[k] = make_a_frag(sH, 128, k << 5, lane);
  const int mb = (lane >= 16) ? 8 : 0;
  const int nlo = lane & 15;
  for (int t = wave; t < 8; t += 4) {
    v8f acc = zero_v8f();
#pragma unroll
    for (int k = 0; k < 4; ++k) acc = wmma_bf16(a[k], load_bfrag(pLin1, t, k, 4, lane), acc);
    const int col = (t << 4) + nlo;
    const float bb = cvec[col];
#pragma unroll
    for (int r = 0; r < 8; ++r) sO[(mb + r) * 128 + col] = lrelu(acc[r] + bb);
  }
  __syncthreads();

  // pack sO (fp32) into bf16 A-fragment layout; wave w owns K-chunk w
  {
    const int m = lane & 15;
    const int kb = (lane >> 4) << 3;
    const float* r = sO + m * 128 + (wave << 5);
    bf16_t* o = hs_pk + ((size_t)blockIdx.x * 4 + wave) * 512 + (size_t)lane * 16;
#pragma unroll
    for (int e = 0; e < 8; ++e) o[e] = (bf16_t)r[kb + e];
#pragma unroll
    for (int e = 0; e < 8; ++e) o[8 + e] = (bf16_t)r[16 + kb + e];
  }
}

// -------------------------------------------------------------- fused head
__device__ __forceinline__ void gemm_afrag(const v16bf* af, const bf16_t* __restrict__ Pb,
                                           const float* __restrict__ bias, float* __restrict__ obuf,
                                           int ncols, int wave, int lane) {
  const int mb = (lane >= 16) ? 8 : 0;
  const int nlo = lane & 15;
  for (int t = wave; (t << 4) < ncols; t += 4) {
    v8f acc = zero_v8f();
#pragma unroll
    for (int k = 0; k < 4; ++k) acc = wmma_bf16(af[k], load_bfrag(Pb, t, k, 4, lane), acc);
    const int col = (t << 4) + nlo;
    const float bb = bias[col];
#pragma unroll
    for (int r = 0; r < 8; ++r) obuf[(mb + r) * 128 + col] = gelu_exact(acc[r] + bb);
  }
}

__device__ __forceinline__ void gemm_ldsA_pk(const float* __restrict__ abuf,
                                             const bf16_t* __restrict__ Pb,
                                             const float* __restrict__ bias, float* __restrict__ obuf,
                                             int ncols, int wave, int lane) {
  v16bf a[4];
#pragma unroll
  for (int k = 0; k < 4; ++k) a[k] = make_a_frag(abuf, 128, k << 5, lane);
  gemm_afrag(a, Pb, bias, obuf, ncols, wave, lane);
}

template <int NK>
__device__ __forceinline__ void gemm_small_pk(const float* __restrict__ abuf,
    const bf16_t* __restrict__ Pb, const float* __restrict__ bias, int ncols,
    float* __restrict__ out, int obase, int z, const int* __restrict__ zidx,
    int m0, int M, int wave, int lane) {
  if (wave != 0) return;
  v8f acc = zero_v8f();
#pragma unroll
  for (int k = 0; k < NK; ++k) {
    v16bf a = make_a_frag(abuf, 128, k << 5, lane);
    acc = wmma_bf16(a, load_bfrag(Pb, 0, k, NK, lane), acc);
  }
  const int col = lane & 15;
  const int mb = (lane >= 16) ? 8 : 0;
  if (col < ncols) {
    const float bb = bias[col];
#pragma unroll
    for (int r = 0; r < 8; ++r) {
      int m = m0 + mb + r;
      if (m < M) {
        float msk = (zidx[m] == z) ? 1.0f : 0.0f;
        out[((size_t)z * M + m) * 22 + obase + col] = (acc[r] + bb) * msk;
      }
    }
  }
}

__global__ __launch_bounds__(128) void head_kernel(
    const bf16_t* __restrict__ hs_pk, const int* __restrict__ zidx,
    const bf16_t* __restrict__ pRW1, const float* __restrict__ rb1,
    const bf16_t* __restrict__ pRW2, const float* __restrict__ rb2,
    const bf16_t* __restrict__ pSWg1, const float* __restrict__ sbg1,
    const bf16_t* __restrict__ pSWg2, const float* __restrict__ sbg2,
    const bf16_t* __restrict__ pSWp1, const float* __restrict__ sbp1,
    const bf16_t* __restrict__ pSWp2, const float* __restrict__ sbp2,
    const bf16_t* __restrict__ pSWp3, const float* __restrict__ sbp3,
    const bf16_t* __restrict__ pGWg1, const float* __restrict__ gbg1,
    const bf16_t* __restrict__ pGWg2, const float* __restrict__ gbg2,
    const bf16_t* __restrict__ pGWp1, const float* __restrict__ gbp1,
    const bf16_t* __restrict__ pGWp2, const float* __restrict__ gbp2,
    const bf16_t* __restrict__ pGWp3, const float* __restrict__ gbp3,
    float* __restrict__ out, int M) {
  __shared__ float sB1[16 * 128];
  __shared__ float sB2[16 * 128];
  const int tid = threadIdx.x, wave = tid >> 5, lane = tid & 31;
  const int z = blockIdx.y;
  const int m0 = blockIdx.x * 16;

  // hs A-fragments: direct 32B vector loads from packed layout, reused by 5 GEMM chains
  v16bf ahs[4];
#pragma unroll
  for (int k = 0; k < 4; ++k)
    ahs[k] = *(const v16bf*)(hs_pk + ((size_t)blockIdx.x * 4 + k) * 512 + (size_t)lane * 16);

  // per-z packed-weight slices
  const bf16_t* rW1 = pRW1 + (size_t)z * 8192;    // 4 tiles * 4 k * 512
  const bf16_t* rW2 = pRW2 + (size_t)z * 1024;    // 1 tile * 2 k * 512
  const bf16_t* wg1 = pSWg1 + (size_t)z * 16384;
  const bf16_t* wg2 = pSWg2 + (size_t)z * 2048;
  const bf16_t* wp1 = pSWp1 + (size_t)z * 16384;
  const bf16_t* wp2 = pSWp2 + (size_t)z * 16384;
  const bf16_t* wp3 = pSWp3 + (size_t)z * 2048;
  const bf16_t* vg1 = pGWg1 + (size_t)z * 16384;
  const bf16_t* vg2 = pGWg2 + (size_t)z * 2048;
  const bf16_t* vp1 = pGWp1 + (size_t)z * 16384;
  const bf16_t* vp2 = pGWp2 + (size_t)z * 16384;
  const bf16_t* vp3 = pGWp3 + (size_t)z * 2048;

  // regression: t = gelu(hs@W1^T+b1) [16x64] ; reg -> cols 0..4
  gemm_afrag(ahs, rW1, rb1 + z * 64, sB1, 64, wave, lane);
  __syncthreads();
  gemm_small_pk<2>(sB1, rW2, rb2 + z * 5, 5, out, 0, z, zidx, m0, M, wave, lane);
  __syncthreads();

  // ssfr gate -> cols 5..7
  gemm_afrag(ahs, wg1, sbg1 + z * 128, sB1, 128, wave, lane);
  __syncthreads();
  gemm_small_pk<4>(sB1, wg2, sbg2 + z * 3, 3, out, 5, z, zidx, m0, M, wave, lane);
  __syncthreads();

  // ssfr predictor -> cols 8..11
  gemm_afrag(ahs, wp1, sbp1 + z * 128, sB1, 128, wave, lane);
  __syncthreads();
  gemm_ldsA_pk(sB1, wp2, sbp2 + z * 128, sB2, 128, wave, lane);
  __syncthreads();
  gemm_small_pk<4>(sB2, wp3, sbp3 + z * 4, 4, out, 8, z, zidx, m0, M, wave, lane);
  __syncthreads();

  // gas gate -> cols 12..15
  gemm_afrag(ahs, vg1, gbg1 + z * 128, sB1, 128, wave, lane);
  __syncthreads();
  gemm_small_pk<4>(sB1, vg2, gbg2 + z * 4, 4, out, 12, z, zidx, m0, M, wave, lane);
  __syncthreads();

  // gas predictor -> cols 16..21
  gemm_afrag(ahs, vp1, gbp1 + z * 128, sB1, 128, wave, lane);
  __syncthreads();
  gemm_ldsA_pk(sB1, vp2, gbp2 + z * 128, sB2, 128, wave, lane);
  __syncthreads();
  gemm_small_pk<4>(sB2, vp3, gbp3 + z * 6, 6, out, 16, z, zidx, m0, M, wave, lane);
}

// ------------------------------------------------------------------ launch
extern "C" void kernel_launch(void* const* d_in, const int* in_sizes, int n_in,
                              void* d_out, int out_size, void* d_ws, size_t ws_size,
                              hipStream_t stream) {
  const float* x     = (const float*)d_in[0];
  const int*   ei    = (const int*)d_in[1];
  const float* u     = (const float*)d_in[2];
  const int*   yidx  = (const int*)d_in[3];
  const int*   zidx  = (const int*)d_in[4];
  const float* c0Wl  = (const float*)d_in[5];
  const float* c0b   = (const float*)d_in[6];
  const float* c0Wr  = (const float*)d_in[7];
  const float* csWl  = (const float*)d_in[8];
  const float* csb   = (const float*)d_in[9];
  const float* csWr  = (const float*)d_in[10];
  const float* lng   = (const float*)d_in[11];
  const float* lnb   = (const float*)d_in[12];
  const float* linuW = (const float*)d_in[13];
  const float* linub = (const float*)d_in[14];
  const float* lin1W = (const float*)d_in[15];
  const float* lin1b = (const float*)d_in[16];
  const float* regW1 = (const float*)d_in[17];
  const float* regb1 = (const float*)d_in[18];
  const float* regW2 = (const float*)d_in[19];
  const float* regb2 = (const float*)d_in[20];
  const float* sWg1 = (const float*)d_in[21];
  const float* sbg1 = (const float*)d_in[22];
  const float* sWg2 = (const float*)d_in[23];
  const float* sbg2 = (const float*)d_in[24];
  const float* sWp1 = (const float*)d_in[25];
  const float* sbp1 = (const float*)d_in[26];
  const float* sWp2 = (const float*)d_in[27];
  const float* sbp2 = (const float*)d_in[28];
  const float* sWp3 = (const float*)d_in[29];
  const float* sbp3 = (const float*)d_in[30];
  const float* gWg1 = (const float*)d_in[31];
  const float* gbg1 = (const float*)d_in[32];
  const float* gWg2 = (const float*)d_in[33];
  const float* gbg2 = (const float*)d_in[34];
  const float* gWp1 = (const float*)d_in[35];
  const float* gbp1 = (const float*)d_in[36];
  const float* gWp2 = (const float*)d_in[37];
  const float* gbp2 = (const float*)d_in[38];
  const float* gWp3 = (const float*)d_in[39];
  const float* gbp3 = (const float*)d_in[40];

  const int N = in_sizes[0] / 64;
  const int E = in_sizes[1] / 2;
  const int M = in_sizes[3];
  const int mtiles = (M + 15) / 16;
  float* out = (float*)d_out;

  char* ws = (char*)d_ws;
  size_t off = 0;
  auto takeb = [&](size_t bytes) -> void* {
    void* p = ws + off;
    off = (off + bytes + 255) & ~(size_t)255;
    return p;
  };
  float* deg  = (float*)takeb((size_t)N * 4);
  float* agg  = (float*)takeb((size_t)N * 128 * 4);
  float* hA   = (float*)takeb((size_t)N * 128 * 4);
  float* hB   = (float*)takeb((size_t)N * 128 * 4);
  float* cvec = (float*)takeb(512);
  bf16_t* hs_pk = (bf16_t*)takeb((size_t)mtiles * 2048 * 2);

  auto pksz = [](int nOut, int K) { return (size_t)((nOut + 15) / 16) * (K / 32) * 512; };
  bf16_t* pk_c0Wl = (bf16_t*)takeb(pksz(128, 64) * 2);
  bf16_t* pk_c0Wr = (bf16_t*)takeb(pksz(128, 64) * 2);
  bf16_t* pk_csWl = (bf16_t*)takeb(3 * pksz(128, 128) * 2);
  bf16_t* pk_csWr = (bf16_t*)takeb(3 * pksz(128, 128) * 2);
  bf16_t* pk_lin1 = (bf16_t*)takeb(pksz(128, 128) * 2);
  bf16_t* pk_rW1  = (bf16_t*)takeb(9 * pksz(64, 128) * 2);
  bf16_t* pk_rW2  = (bf16_t*)takeb(9 * pksz(5, 64) * 2);
  bf16_t* pk_sg1  = (bf16_t*)takeb(9 * pksz(128, 128) * 2);
  bf16_t* pk_sg2  = (bf16_t*)takeb(9 * pksz(3, 128) * 2);
  bf16_t* pk_sp1  = (bf16_t*)takeb(9 * pksz(128, 128) * 2);
  bf16_t* pk_sp2  = (bf16_t*)takeb(9 * pksz(128, 128) * 2);
  bf16_t* pk_sp3  = (bf16_t*)takeb(9 * pksz(4, 128) * 2);
  bf16_t* pk_gg1  = (bf16_t*)takeb(9 * pksz(128, 128) * 2);
  bf16_t* pk_gg2  = (bf16_t*)takeb(9 * pksz(4, 128) * 2);
  bf16_t* pk_gp1  = (bf16_t*)takeb(9 * pksz(128, 128) * 2);
  bf16_t* pk_gp2  = (bf16_t*)takeb(9 * pksz(128, 128) * 2);
  bf16_t* pk_gp3  = (bf16_t*)takeb(9 * pksz(6, 128) * 2);

  auto pack = [&](const float* W, bf16_t* P, int nOut, int K, int ldw,
                  long long sWz, long long sPz, int Zc) {
    int total = ((nOut + 15) / 16) * (K / 32) * 32;
    dim3 g((total + 255) / 256, Zc);
    pack_b_kernel<<<g, 256, 0, stream>>>(W, P, nOut, K, ldw, sWz, sPz);
  };
  pack(c0Wl, pk_c0Wl, 128, 64, 64, 0, 0, 1);
  pack(c0Wr, pk_c0Wr, 128, 64, 64, 0, 0, 1);
  pack(csWl, pk_csWl, 128, 128, 128, 16384, 16384, 3);
  pack(csWr, pk_csWr, 128, 128, 128, 16384, 16384, 3);
  pack(lin1W, pk_lin1, 128, 128, 256, 0, 0, 1);
  pack(regW1, pk_rW1, 64, 128, 128, 8192, 8192, 9);
  pack(regW2, pk_rW2, 5, 64, 64, 320, 1024, 9);
  pack(sWg1, pk_sg1, 128, 128, 128, 16384, 16384, 9);
  pack(sWg2, pk_sg2, 3, 128, 128, 384, 2048, 9);
  pack(sWp1, pk_sp1, 128, 128, 128, 16384, 16384, 9);
  pack(sWp2, pk_sp2, 128, 128, 128, 16384, 16384, 9);
  pack(sWp3, pk_sp3, 4, 128, 128, 512, 2048, 9);
  pack(gWg1, pk_gg1, 128, 128, 128, 16384, 16384, 9);
  pack(gWg2, pk_gg2, 4, 128, 128, 512, 2048, 9);
  pack(gWp1, pk_gp1, 128, 128, 128, 16384, 16384, 9);
  pack(gWp2, pk_gp2, 128, 128, 128, 16384, 16384, 9);
  pack(gWp3, pk_gp3, 6, 128, 128, 768, 2048, 9);

  zero_kernel<<<(N + 255) / 256, 256, 0, stream>>>(deg, N);
  degree_kernel<<<(E + 255) / 256, 256, 0, stream>>>(ei, deg, E);
  prep_kernel<<<1, 128, 0, stream>>>(u, linuW, linub, lin1W, lin1b, cvec);

  const float* hin = x;
  float* hout = hA;
  for (int i = 0; i < 4; ++i) {
    const int Kin = (i == 0) ? 64 : 128;
    const long long an = (long long)N * Kin;
    zero_kernel<<<(int)((an + 255) / 256), 256, 0, stream>>>(agg, an);
    const long long sth = (long long)E * (Kin / 4);
    scatter_kernel<<<(int)((sth + 255) / 256), 256, 0, stream>>>(hin, ei, agg, E, Kin);
    if (i == 0) {
      gnn_layer_kernel<64><<<(N + 15) / 16, 128, 0, stream>>>(
          hin, agg, deg, pk_c0Wl, c0b, pk_c0Wr, lng, lnb, hout, N);
    } else {
      gnn_layer_kernel<128><<<(N + 15) / 16, 128, 0, stream>>>(
          hin, agg, deg, pk_csWl + (size_t)(i - 1) * 16384, csb + (size_t)(i - 1) * 128,
          pk_csWr + (size_t)(i - 1) * 16384, lng + i * 128, lnb + i * 128, hout, N);
    }
    hin = hout;
    hout = (hout == hA) ? hB : hA;
  }

  hs_kernel<<<mtiles, 128, 0, stream>>>(hin, yidx, pk_lin1, cvec, hs_pk, M);

  dim3 hg(mtiles, 9);
  head_kernel<<<hg, 128, 0, stream>>>(hs_pk, zidx,
      pk_rW1, regb1, pk_rW2, regb2,
      pk_sg1, sbg1, pk_sg2, sbg2, pk_sp1, sbp1, pk_sp2, sbp2, pk_sp3, sbp3,
      pk_gg1, gbg1, pk_gg2, gbg2, pk_gp1, gbp1, pk_gp2, gbp2, pk_gp3, gbp3,
      out, M);
}
